// SimplifiedMamba2Block_30623116820623
// MI455X (gfx1250) — matrix operand order repaired and checked by reading, not compiled, not run on hardware
//
#include <hip/hip_runtime.h>
#include <hip/hip_bf16.h>
#include <stdint.h>

// ---------------------------------------------------------------------------
// SimplifiedMamba2Block for MI455X (gfx1250, wave32, WMMA)
//   LN -> GEMM1 (bf16 WMMA, async-LDS staged) -> fwd/bwd SSM scan + SiLU gate
//      -> GEMM2 (bf16 WMMA, async-LDS staged)
// ---------------------------------------------------------------------------

#define D_MODEL 1024
#define D_STATE 16
#define D_INNER 2048
#define BATCH   4
#define SEQ     2048
#define NTOK    (BATCH * SEQ)   // 8192 tokens
#define LN_EPS  1e-5f

typedef __attribute__((ext_vector_type(16))) __bf16 v16bf;
typedef __attribute__((ext_vector_type(8)))  float  v8f;

static __device__ __forceinline__ unsigned short f2bf(float f) {
    union { float f; uint32_t u; } v; v.f = f;
    uint32_t r = v.u + 0x7fffu + ((v.u >> 16) & 1u);   // round-to-nearest-even
    return (unsigned short)(r >> 16);
}

// one per-lane 16-byte async copy global -> LDS (ASYNCcnt-tracked)
static __device__ __forceinline__ void async_copy16(uint32_t lds_off,
                                                    const void* gptr) {
    asm volatile("global_load_async_to_lds_b128 %0, %1, off"
                 :: "v"(lds_off), "v"((uint64_t)gptr)
                 : "memory");
}

// -------------------------- weight conversion ------------------------------
__global__ void cvt_f32_to_bf16(const float* __restrict__ in,
                                unsigned short* __restrict__ out, int n) {
    int i = blockIdx.x * blockDim.x + threadIdx.x;
    if (i < n) out[i] = f2bf(in[i]);
}

// ------------------------------ layernorm ----------------------------------
// one 256-thread block (8 waves) per token; D_MODEL = 1024 = 4 * 256
__global__ void ln_kernel(const float* __restrict__ x,
                          const float* __restrict__ w,
                          const float* __restrict__ b,
                          unsigned short* __restrict__ out) {
    __shared__ float s_sum[8];
    __shared__ float s_sq[8];
    const int token = blockIdx.x;
    const int t = threadIdx.x;
    const float* xr = x + (size_t)token * D_MODEL;

    float vals[4];
    float lsum = 0.f, lsq = 0.f;
#pragma unroll
    for (int j = 0; j < 4; ++j) {
        float v = xr[t + j * 256];
        vals[j] = v; lsum += v; lsq += v * v;
    }
#pragma unroll
    for (int off = 16; off > 0; off >>= 1) {
        lsum += __shfl_down(lsum, off, 32);
        lsq  += __shfl_down(lsq,  off, 32);
    }
    const int lane = t & 31, wv = t >> 5;
    if (lane == 0) { s_sum[wv] = lsum; s_sq[wv] = lsq; }
    __syncthreads();
    float tot = 0.f, totsq = 0.f;
#pragma unroll
    for (int i = 0; i < 8; ++i) { tot += s_sum[i]; totsq += s_sq[i]; }
    const float mu   = tot * (1.f / D_MODEL);
    const float var  = totsq * (1.f / D_MODEL) - mu * mu;
    const float rstd = rsqrtf(var + LN_EPS);

    unsigned short* orow = out + (size_t)token * D_MODEL;
#pragma unroll
    for (int j = 0; j < 4; ++j) {
        int c = t + j * 256;
        orow[c] = f2bf((vals[j] - mu) * rstd * w[c] + b[c]);
    }
}

// ------------------------------ WMMA GEMM ----------------------------------
// C(MxN, f32) = A(MxK, bf16 row-major) * B^T, B stored (NxK, bf16 row-major).
// Block = 256 threads = 8 waves; block tile 128x64; wave tile 32x32
// (2x2 x 16x16 frags).  A/B panels double-buffered in LDS, filled with
// global_load_async_to_lds_b128, consumed via ds_load_b128.
// M%128==0, N%64==0, K%32==0 guaranteed by problem dims.
#define ABUF (128 * 32)   // bf16 elems per A buffer (8 KB)
#define BBUF (64 * 32)    // bf16 elems per B buffer (4 KB)

__global__ void __launch_bounds__(256, 2)
gemm_bf16_wmma(const unsigned short* __restrict__ A,
               const unsigned short* __restrict__ B,
               float* __restrict__ C,
               int M, int N, int K) {
    __shared__ alignas(16) unsigned short sA[2 * ABUF];
    __shared__ alignas(16) unsigned short sB[2 * BBUF];

    const int tid  = threadIdx.x;
    const int lane = tid & 31;
    const int wave = tid >> 5;
    const int wm = wave & 3;             // 4 waves along M
    const int wn = wave >> 2;            // 2 waves along N
    const int mblk = blockIdx.y * 128;
    const int nblk = blockIdx.x * 64;

    const int rr = lane & 15;
    // A fragment (16x32 bf16): lanes 0-15 take K chunks {0..7,16..23},
    // lanes 16-31 take {8..15,24..31}
    const int ka = (lane >> 4) << 3;     // 0 or 8
    // B fragment (32x16 bf16, B^T rows contiguous): lane l holds column
    // N=l%16, K chunks {0..7,16..23} (lanes 0-15) / {8..15,24..31} (lanes 16-31)
    const int kb = (lane >> 4) << 3;     // 0 or 8

    // LDS byte offsets (generic-pointer low 32 bits == LDS offset)
    const uint32_t sA_off = (uint32_t)(uintptr_t)&sA[0];
    const uint32_t sB_off = (uint32_t)(uintptr_t)&sB[0];

    // per-thread copy assignments: 3 x b128 per K-chunk
    const int arow0 = tid >> 2,         aq0 = tid & 3;           // A rows 0..63
    const int arow1 = (tid + 256) >> 2, aq1 = tid & 3;           // A rows 64..127
    const int brow  = tid >> 2,         bq  = tid & 3;           // B rows 0..63

    const unsigned short* gA0 = A + (size_t)(mblk + arow0) * K + aq0 * 8;
    const unsigned short* gA1 = A + (size_t)(mblk + arow1) * K + aq1 * 8;
    const unsigned short* gB0 = B + (size_t)(nblk + brow)  * K + bq * 8;
    const uint32_t lA0 = sA_off + (uint32_t)(arow0 * 64 + aq0 * 16);
    const uint32_t lA1 = sA_off + (uint32_t)(arow1 * 64 + aq1 * 16);
    const uint32_t lB0 = sB_off + (uint32_t)(brow * 64 + bq * 16);

    // prologue: stage chunk 0 into buffer 0
    async_copy16(lA0, gA0);
    async_copy16(lA1, gA1);
    async_copy16(lB0, gB0);

    v8f acc00 = {}, acc01 = {}, acc10 = {}, acc11 = {};

    for (int kc = 0, c = 0; kc < K; kc += 32, ++c) {
        const int buf = c & 1;
        const int kn = kc + 32;
        if (kn < K) {
            const uint32_t boff = (uint32_t)((buf ^ 1) ? 1 : 0);
            async_copy16(lA0 + boff * (ABUF * 2), gA0 + kn);
            async_copy16(lA1 + boff * (ABUF * 2), gA1 + kn);
            async_copy16(lB0 + boff * (BBUF * 2), gB0 + kn);
            // pull the chunk after that toward GL2 (global_prefetch_b8)
            __builtin_prefetch(gA0 + kn + 32, 0, 0);
            __builtin_prefetch(gB0 + kn + 32, 0, 0);
            asm volatile("s_wait_asynccnt 0x3" ::: "memory"); // current chunk done
        } else {
            asm volatile("s_wait_asynccnt 0x0" ::: "memory");
        }
        __syncthreads();   // chunk visible block-wide

        const unsigned short* pA = sA + buf * ABUF;
        const unsigned short* pB = sB + buf * BBUF;

        union { uint4 q[2]; v16bf v; } fa0, fa1, fb0, fb1;
        const unsigned short* a0p = pA + (wm * 32 + rr) * 32 + ka;
        fa0.q[0] = *(const uint4*)(a0p);
        fa0.q[1] = *(const uint4*)(a0p + 16);
        const unsigned short* a1p = a0p + 16 * 32;
        fa1.q[0] = *(const uint4*)(a1p);
        fa1.q[1] = *(const uint4*)(a1p + 16);
        const unsigned short* b0p = pB + (wn * 32 + rr) * 32 + kb;
        fb0.q[0] = *(const uint4*)(b0p);
        fb0.q[1] = *(const uint4*)(b0p + 16);
        const unsigned short* b1p = b0p + 16 * 32;
        fb1.q[0] = *(const uint4*)(b1p);
        fb1.q[1] = *(const uint4*)(b1p + 16);

        acc00 = __builtin_amdgcn_wmma_f32_16x16x32_bf16(
                    false, fa0.v, false, fb0.v, (short)0, acc00, false, false);
        acc01 = __builtin_amdgcn_wmma_f32_16x16x32_bf16(
                    false, fa0.v, false, fb1.v, (short)0, acc01, false, false);
        acc10 = __builtin_amdgcn_wmma_f32_16x16x32_bf16(
                    false, fa1.v, false, fb0.v, (short)0, acc10, false, false);
        acc11 = __builtin_amdgcn_wmma_f32_16x16x32_bf16(
                    false, fa1.v, false, fb1.v, (short)0, acc11, false, false);

        __syncthreads();   // everyone done reading buf before it is refilled
    }

    // C/D layout: VGPR r -> row (r + 8*(lane>>4)), col lane%16
    const int m0 = mblk + wm * 32;
    const int n0 = nblk + wn * 32;
    const int crow = (lane >> 4) << 3;
    const int ccol = lane & 15;
#pragma unroll
    for (int r = 0; r < 8; ++r) {
        C[(size_t)(m0 + crow + r)      * N + n0 + ccol]      = acc00[r];
        C[(size_t)(m0 + crow + r)      * N + n0 + 16 + ccol] = acc01[r];
        C[(size_t)(m0 + 16 + crow + r) * N + n0 + ccol]      = acc10[r];
        C[(size_t)(m0 + 16 + crow + r) * N + n0 + 16 + ccol] = acc11[r];
    }
}

// ---------------------------- SSM scan + gate ------------------------------
// One lane per (batch, inner-channel); 16-state recurrence held in VGPRs.
// Pass 1: forward scan -> yfwd.  Pass 2: backward scan, combine, SiLU gate,
// emit bf16 activation for the output GEMM.
__global__ void __launch_bounds__(256, 4)
scan_kernel(const float* __restrict__ xproj,   // NTOK x (2*D_INNER)
            const float* __restrict__ A_log,   // D_INNER x D_STATE
            const float* __restrict__ Bp,      // D_INNER x D_STATE
            const float* __restrict__ Cp,      // D_INNER x D_STATE
            float* __restrict__ yfwd,          // NTOK x D_INNER
            unsigned short* __restrict__ act)  // NTOK x D_INNER (bf16)
{
    const int gid = blockIdx.x * blockDim.x + threadIdx.x;   // 0..8191
    const int b = gid >> 11;            // / D_INNER
    const int i = gid & (D_INNER - 1);

    float Ar[D_STATE], Br[D_STATE], Cr[D_STATE], h[D_STATE];
#pragma unroll
    for (int s = 0; s < D_STATE; ++s) {
        Ar[s] = -__expf(A_log[i * D_STATE + s]);
        Br[s] = Bp[i * D_STATE + s];
        Cr[s] = Cp[i * D_STATE + s];
        h[s]  = 0.f;
    }

    const size_t rs = 2 * D_INNER;
    const float* xb = xproj + (size_t)b * SEQ * rs + i;            // x_ssm col
    const float* gb = xb + D_INNER;                                // gate col
    float* yb = yfwd + (size_t)b * SEQ * D_INNER + i;
    unsigned short* ab = act + (size_t)b * SEQ * D_INNER + i;

    // forward scan
    for (int t = 0; t < SEQ; ++t) {
        float x = xb[(size_t)t * rs];
        float y = 0.f;
#pragma unroll
        for (int s = 0; s < D_STATE; ++s) {
            h[s] = fmaf(h[s], Ar[s], x * Br[s]);
            y = fmaf(h[s], Cr[s], y);
        }
        yb[(size_t)t * D_INNER] = y;
    }

    // backward scan + combine + SiLU gate
#pragma unroll
    for (int s = 0; s < D_STATE; ++s) h[s] = 0.f;
    for (int t = SEQ - 1; t >= 0; --t) {
        float x = xb[(size_t)t * rs];
        float y = 0.f;
#pragma unroll
        for (int s = 0; s < D_STATE; ++s) {
            h[s] = fmaf(h[s], Ar[s], x * Br[s]);
            y = fmaf(h[s], Cr[s], y);
        }
        float g    = gb[(size_t)t * rs];
        float silu = g / (1.f + __expf(-g));
        float o    = 0.5f * (yb[(size_t)t * D_INNER] + y) * silu;
        ab[(size_t)t * D_INNER] = f2bf(o);
    }
}

// ------------------------------- launcher ----------------------------------
extern "C" void kernel_launch(void* const* d_in, const int* in_sizes, int n_in,
                              void* d_out, int out_size, void* d_ws, size_t ws_size,
                              hipStream_t stream) {
    (void)in_sizes; (void)n_in; (void)out_size; (void)ws_size;

    const float* x     = (const float*)d_in[0];
    const float* W_in  = (const float*)d_in[1];
    const float* A_log = (const float*)d_in[2];
    const float* B_par = (const float*)d_in[3];
    const float* C_par = (const float*)d_in[4];
    const float* W_out = (const float*)d_in[5];
    const float* ln_w  = (const float*)d_in[6];
    const float* ln_b  = (const float*)d_in[7];
    float* out = (float*)d_out;

    // workspace carve-out (all buffers fully written before read, every call)
    char* ws = (char*)d_ws;
    size_t off = 0;
    auto carve = [&](size_t bytes) -> void* {
        off = (off + 255) & ~(size_t)255;
        void* p = ws + off;
        off += bytes;
        return p;
    };
    unsigned short* xln_bf   = (unsigned short*)carve((size_t)NTOK * D_MODEL * 2);
    unsigned short* w_in_bf  = (unsigned short*)carve((size_t)2 * D_INNER * D_MODEL * 2);
    unsigned short* w_out_bf = (unsigned short*)carve((size_t)D_MODEL * D_INNER * 2);
    float*          xproj    = (float*)carve((size_t)NTOK * 2 * D_INNER * 4);
    float*          yfwd     = (float*)carve((size_t)NTOK * D_INNER * 4);
    unsigned short* act_bf   = (unsigned short*)carve((size_t)NTOK * D_INNER * 2);

    // 1) weights -> bf16
    {
        int n1 = 2 * D_INNER * D_MODEL;            // W_in: 4096x1024
        cvt_f32_to_bf16<<<(n1 + 255) / 256, 256, 0, stream>>>(W_in, w_in_bf, n1);
        int n2 = D_MODEL * D_INNER;                // W_out: 1024x2048
        cvt_f32_to_bf16<<<(n2 + 255) / 256, 256, 0, stream>>>(W_out, w_out_bf, n2);
    }

    // 2) layernorm -> bf16 activations
    ln_kernel<<<NTOK, 256, 0, stream>>>(x, ln_w, ln_b, xln_bf);

    // 3) GEMM1: xproj(8192x4096) = xln(8192x1024) @ W_in^T
    {
        dim3 grid((2 * D_INNER) / 64, NTOK / 128);
        gemm_bf16_wmma<<<grid, 256, 0, stream>>>(xln_bf, w_in_bf, xproj,
                                                 NTOK, 2 * D_INNER, D_MODEL);
    }

    // 4) fwd+bwd scans, combine, SiLU gate -> bf16 activations
    scan_kernel<<<(BATCH * D_INNER) / 256, 256, 0, stream>>>(
        xproj, A_log, B_par, C_par, yfwd, act_bf);

    // 5) GEMM2: out(8192x1024) = act(8192x2048) @ W_out^T
    {
        dim3 grid(D_MODEL / 64, NTOK / 128);
        gemm_bf16_wmma<<<grid, 256, 0, stream>>>(act_bf, w_out_bf, out,
                                                 NTOK, D_MODEL, D_INNER);
    }
}
